// DepthwiseKPConv_12549894439612
// MI455X (gfx1250) — matrix operand order repaired and checked by reading, not compiled, not run on hardware
//
#include <hip/hip_runtime.h>
#include <hip/hip_bf16.h>

typedef __attribute__((ext_vector_type(16))) _Float16 v16h;
typedef __attribute__((ext_vector_type(8)))  float    v8f;

#define NUM_KERNEL 15
#define DCH 64
#define KNB 32
// SCALE = 0.3^2*2 + 1e-10
#define INV_SCALE (1.0f / (0.3f * 0.3f * 2.0f + 1e-10f))
#define BN_EPS 1e-5f

// One wave32 per point. A = corr (16x32 f16, row 15 zero-padded),
// B = xj tile (32x16 f16), D = agg tile (16x16 f32) via v_wmma_f32_16x16x32_f16.
__global__ void kpconv_wmma_kernel(const float* __restrict__ pj,
                                   const float* __restrict__ xj,
                                   const float* __restrict__ kernel_point,
                                   const float* __restrict__ w_dw,
                                   const float* __restrict__ b_dw,
                                   float* __restrict__ y_out,
                                   float* __restrict__ partials,
                                   int N, int totalWaves) {
    __shared__ float kp_s[NUM_KERNEL * 3];
    __shared__ float w_s[DCH * 16];   // w_dw padded [c][16], m=15 -> 0
    __shared__ float b_s[DCH];

    for (int i = threadIdx.x; i < NUM_KERNEL * 3; i += blockDim.x) kp_s[i] = kernel_point[i];
    for (int i = threadIdx.x; i < DCH * 16; i += blockDim.x) {
        int c = i >> 4, m = i & 15;
        w_s[i] = (m < NUM_KERNEL) ? w_dw[c * NUM_KERNEL + m] : 0.0f;
    }
    for (int i = threadIdx.x; i < DCH; i += blockDim.x) b_s[i] = b_dw[i];
    __syncthreads();

    const int lane   = threadIdx.x & 31;
    const int wave   = (blockIdx.x * blockDim.x + threadIdx.x) >> 5;
    const int laneHi = lane >> 4;      // which half-wave
    const int cl     = lane & 15;      // column within tile / A row
    const int m      = lane & 15;      // kernel-point row this lane owns in A
    const bool mvalid = (m < NUM_KERNEL);
    const float kpx = mvalid ? kp_s[m * 3 + 0] : 0.0f;
    const float kpy = mvalid ? kp_s[m * 3 + 1] : 0.0f;
    const float kpz = mvalid ? kp_s[m * 3 + 2] : 0.0f;

    float sum_t[4] = {0.f, 0.f, 0.f, 0.f};
    float sq_t[4]  = {0.f, 0.f, 0.f, 0.f};

    for (int n = wave; n < N; n += totalWaves) {
        const float* pjn = pj + (size_t)n * KNB * 3;
        const float* xjn = xj + (size_t)n * KNB * DCH;

        // prefetch next point's xj stream (emits global_prefetch_b8)
        if (n + totalWaves < N)
            __builtin_prefetch(xj + (size_t)(n + totalWaves) * KNB * DCH + lane * 64, 0, 1);

        // lane = neighbor k: load offset, max-|p| over wave, normalize
        float px = pjn[lane * 3 + 0];
        float py = pjn[lane * 3 + 1];
        float pz = pjn[lane * 3 + 2];
        float l2 = sqrtf(px * px + py * py + pz * pz);
        float mx = l2;
        #pragma unroll
        for (int off = 16; off > 0; off >>= 1) mx = fmaxf(mx, __shfl_xor(mx, off, 32));
        float inv = 1.0f / (mx + 1e-10f);
        px *= inv; py *= inv; pz *= inv;

        // Build A (corr) directly in the 16-bit 16x32 A-matrix VGPR layout:
        // lanes 0-15 row m: h=0..7 -> K=h, h=8..15 -> K=h+8
        // lanes 16-31 row m: h=0..7 -> K=h+8, h=8..15 -> K=h+16
        v16h a;
        #pragma unroll
        for (int h = 0; h < 16; ++h) {
            int k = h + ((h & 8) ? 8 : 0) + (laneHi ? 8 : 0);
            float qx = __shfl(px, k, 32);
            float qy = __shfl(py, k, 32);
            float qz = __shfl(pz, k, 32);
            float dx = kpx - qx, dy = kpy - qy, dz = kpz - qz;
            float sq = dx * dx + dy * dy + dz * dz;
            float corr = mvalid ? __expf(-sq * INV_SCALE) : 0.0f;
            a[h] = (_Float16)corr;
        }

        // 4 column tiles of C=64
        #pragma unroll
        for (int t = 0; t < 4; ++t) {
            // B 32x16 layout: lane = N (column), same K order as A: h=0..7 -> K=h+8*laneHi, h=8..15 -> K=h+8+8*laneHi
            v16h b;
            #pragma unroll
            for (int h = 0; h < 16; ++h) {
                int kk = h + ((h & 8) ? 8 : 0) + (laneHi ? 8 : 0);
                b[h] = (_Float16)xjn[kk * DCH + t * 16 + cl];
            }
            v8f acc = {};
            acc = __builtin_amdgcn_wmma_f32_16x16x32_f16(
                false, a, false, b, (short)0, acc, false, false);

            // D layout: vgpr r, lanes 0-15: m=r; lanes 16-31: m=8+r (same column cl)
            int c = t * 16 + cl;
            float part = 0.0f;
            #pragma unroll
            for (int r = 0; r < 8; ++r) {
                int mm = r + (laneHi ? 8 : 0);
                part += acc[r] * w_s[c * 16 + mm];   // w_s[.][15] == 0 handles padding
            }
            part += __shfl_xor(part, 16, 32);        // combine m=0..7 and m=8..15 halves
            float yv = part + b_s[c];
            if (!laneHi) y_out[(size_t)n * DCH + c] = yv;
            sum_t[t] += yv;
            sq_t[t]  += yv * yv;
        }
    }

    // per-wave BN partials (fixed order -> deterministic), lanes 0-15 only
    if (!laneHi) {
        #pragma unroll
        for (int t = 0; t < 4; ++t) {
            partials[(size_t)wave * 128 + t * 16 + cl]      = sum_t[t];
            partials[(size_t)wave * 128 + 64 + t * 16 + cl] = sq_t[t];
        }
    }
}

__global__ void kpconv_bn_stats_kernel(const float* __restrict__ partials,
                                       float* __restrict__ stats,
                                       int W, int N) {
    int c = threadIdx.x;
    if (c < DCH) {
        float s = 0.0f, q = 0.0f;
        for (int w = 0; w < W; ++w) {
            s += partials[(size_t)w * 128 + c];
            q += partials[(size_t)w * 128 + 64 + c];
        }
        float invN = 1.0f / (float)N;
        float mean = s * invN;
        float var  = q * invN - mean * mean;
        stats[c]       = mean;
        stats[64 + c]  = rsqrtf(var + BN_EPS);
    }
}

__global__ void kpconv_bn_apply_kernel(const float* __restrict__ y,
                                       const float* __restrict__ stats,
                                       const float* __restrict__ gamma,
                                       const float* __restrict__ beta,
                                       float* __restrict__ out, int total) {
    int i = blockIdx.x * blockDim.x + threadIdx.x;
    if (i < total) {
        int c = i & (DCH - 1);
        float v = (y[i] - stats[c]) * stats[64 + c] * gamma[c] + beta[c];
        out[i] = fmaxf(v, 0.0f);
    }
}

extern "C" void kernel_launch(void* const* d_in, const int* in_sizes, int n_in,
                              void* d_out, int out_size, void* d_ws, size_t ws_size,
                              hipStream_t stream) {
    // inputs: p[0](unused), pj[1], x[2](unused), xj[3], kernel_point[4],
    //         w_dw[5], b_dw[6], gamma[7], beta[8]
    const float* pj    = (const float*)d_in[1];
    const float* xj    = (const float*)d_in[3];
    const float* kp    = (const float*)d_in[4];
    const float* w_dw  = (const float*)d_in[5];
    const float* b_dw  = (const float*)d_in[6];
    const float* gamma = (const float*)d_in[7];
    const float* beta  = (const float*)d_in[8];
    const int N = in_sizes[0] / 3;

    const int threads = 256;                 // 8 wave32 per block
    const int blocks  = 512;
    const int W = blocks * (threads / 32);   // total waves

    float* ws       = (float*)d_ws;
    float* y        = ws;                          // N*64
    float* partials = y + (size_t)N * DCH;         // W*128
    float* stats    = partials + (size_t)W * 128;  // 128

    kpconv_wmma_kernel<<<blocks, threads, 0, stream>>>(
        pj, xj, kp, w_dw, b_dw, y, partials, N, W);
    kpconv_bn_stats_kernel<<<1, 64, 0, stream>>>(partials, stats, W, N);
    int total = N * DCH;
    kpconv_bn_apply_kernel<<<(total + 255) / 256, 256, 0, stream>>>(
        y, stats, gamma, beta, (float*)d_out, total);
}
